// Dynamic_QLayer_Block_89859305767554
// MI455X (gfx1250) — matrix operand order repaired and checked by reading, not compiled, not run on hardware
//
#include <hip/hip_runtime.h>
#include <hip/hip_bf16.h>
#include <math.h>

// ---------------------------------------------------------------------------
// ResMLP-style block with per-tensor 8-bit fake-quant weights, for gfx1250.
// f16 WMMA (V_WMMA_F32_16X16X32_F16), fp32 accumulate, fused epilogues.
// Register-blocked 2x2 (tokenmix) / 2x1 (channelmix) tiles per wave to cut
// L2 weight-fragment traffic; 32-row channel-MLP tile in dynamic LDS.
// ---------------------------------------------------------------------------

typedef __attribute__((ext_vector_type(16))) _Float16 v16h;
typedef __attribute__((ext_vector_type(8)))  _Float16 v8h;
typedef __attribute__((ext_vector_type(8)))  float    v8f;

#define B_  256
#define N_  196
#define D_  384
#define H_  1536
#define NPAD_M 208   // padded W_attn rows (m): 13*16
#define NPAD_K 224   // padded W_attn cols (n): 7*32

#define X2S_STRIDE 392    // 784 B rows: 16B multiple, bank-spread
#define H1S_STRIDE 1544   // 3088 B rows: 16B multiple, bank-spread
#define CM_ROWS 32
#define CM_LDS_BYTES (CM_ROWS * X2S_STRIDE * 2 + CM_ROWS * H1S_STRIDE * 2)

__device__ __forceinline__ v16h cat16(v8h lo, v8h hi) {
  return __builtin_shufflevector(lo, hi, 0,1,2,3,4,5,6,7,8,9,10,11,12,13,14,15);
}
__device__ __forceinline__ v8f wmma_f16(v16h a, v16h b, v8f c) {
  return __builtin_amdgcn_wmma_f32_16x16x32_f16(false, a, false, b, (short)0, c, false, false);
}
__device__ __forceinline__ float gelu_tanh(float v) {
  const float k = 0.7978845608028654f;
  float t = tanhf(k * (v + 0.044715f * v * v * v));
  return 0.5f * v * (1.0f + t);
}

// ---------------------------- weight quantization ---------------------------

__global__ void k_init(int* absmax) {
  if (threadIdx.x < 3) absmax[threadIdx.x] = 0;
}

__global__ __launch_bounds__(256) void k_absmax(const float* __restrict__ src,
                                                int n, int* __restrict__ slot) {
  __shared__ float red[256];
  float m = 0.0f;
  for (int i = blockIdx.x * blockDim.x + threadIdx.x; i < n; i += gridDim.x * blockDim.x)
    m = fmaxf(m, fabsf(src[i]));
  red[threadIdx.x] = m;
  __syncthreads();
  for (int s = 128; s > 0; s >>= 1) {
    if (threadIdx.x < s) red[threadIdx.x] = fmaxf(red[threadIdx.x], red[threadIdx.x + s]);
    __syncthreads();
  }
  if (threadIdx.x == 0) atomicMax(slot, __float_as_int(red[0]));  // vals >= 0
}

// quantize src[rows][cols] -> dst[.][pcols] f16 (zero pad out of range)
__global__ __launch_bounds__(256) void k_quant(const float* __restrict__ src,
                                               _Float16* __restrict__ dst,
                                               int rows, int cols, int pcols, int total,
                                               const int* __restrict__ slot,
                                               const int* __restrict__ bitsPtr) {
  int i = blockIdx.x * 256 + threadIdx.x;
  if (i >= total) return;
  int r = i / pcols, c = i % pcols;
  float out = 0.0f;
  if (r < rows && c < cols) {
    float qmax  = (float)((1 << (*bitsPtr - 1)) - 1);
    float mx    = __int_as_float(*slot);
    float scale = fmaxf(mx / qmax, 1e-8f);
    float w = src[r * cols + c];
    float q = rintf(w / scale);               // round-half-even, matches jnp.round
    q = fminf(fmaxf(q, -qmax), qmax);
    out = q * scale;
  }
  dst[i] = (_Float16)out;
}

// ------------------------------- token mixing -------------------------------
// out[b,m,d] = gamma1[d]*( sum_n Wq[m,n]*aff1(x)[b,n,d] + b_attn[m] ) + x[b,m,d]
// grid: (D/32, B); block 256 (8 waves). Per wave: 2 M-tiles x 2 d-tiles so one
// A fragment (Wq rows, d-independent) feeds 2 WMMAs and one staged B fragment
// feeds 2 WMMAs.

__global__ __launch_bounds__(256) void k_tokenmix(
    const float* __restrict__ x, const float* __restrict__ alpha1,
    const float* __restrict__ beta1, const _Float16* __restrict__ Wq,
    const float* __restrict__ b_attn, const float* __restrict__ gamma1,
    float* __restrict__ out)
{
  __shared__ __align__(16) _Float16 hsT[32][40];  // [d][n] chunk, padded stride
  const int dbase = blockIdx.x * 32;
  const int b     = blockIdx.y;
  const int tid   = threadIdx.x;
  const int lane  = tid & 31;
  const int wid   = tid >> 5;
  const int nsel  = lane & 15;     // M row / N col selector
  const int khalf = lane >> 4;     // lane-half -> K sub-block

  v8f acc[2][2] = {};              // [m-tile][d-tile]
  const int mt0 = wid;
  const int mt1 = wid + 8;         // 13 M-tiles over 8 waves (uniform per wave)

  for (int kc = 0; kc < NPAD_K / 32; ++kc) {
    // stage transposed aff1(x) chunk: hsT[d][n_local], fully coalesced in d
    for (int idx = tid; idx < 1024; idx += 256) {
      int d  = idx & 31;
      int nl = idx >> 5;
      int n  = kc * 32 + nl;
      int dg = dbase + d;
      float v = 0.0f;
      if (n < N_) v = alpha1[dg] * x[(b * N_ + n) * D_ + dg] + beta1[dg];
      hsT[d][nl] = (_Float16)v;
    }
    __syncthreads();

    // B operands: lane -> N (= local d), K = khalf*16 + e (contiguous LDS row)
    v16h bm0 = cat16(*(const v8h*)&hsT[nsel][khalf * 16],
                     *(const v8h*)&hsT[nsel][khalf * 16 + 8]);
    v16h bm1 = cat16(*(const v8h*)&hsT[16 + nsel][khalf * 16],
                     *(const v8h*)&hsT[16 + nsel][khalf * 16 + 8]);

    {
      int row = mt0 * 16 + nsel;
      const _Float16* ap = Wq + row * NPAD_K + kc * 32 + khalf * 16;
      v16h am = cat16(*(const v8h*)ap, *(const v8h*)(ap + 8));
      acc[0][0] = wmma_f16(am, bm0, acc[0][0]);
      acc[0][1] = wmma_f16(am, bm1, acc[0][1]);
    }
    if (mt1 < 13) {  // wave-uniform at runtime: EXEC all-ones inside
      int row = mt1 * 16 + nsel;
      const _Float16* ap = Wq + row * NPAD_K + kc * 32 + khalf * 16;
      v16h am = cat16(*(const v8h*)ap, *(const v8h*)(ap + 8));
      acc[1][0] = wmma_f16(am, bm0, acc[1][0]);
      acc[1][1] = wmma_f16(am, bm1, acc[1][1]);
    }
    __syncthreads();
  }

  for (int t = 0; t < 2; ++t) {
    int mt = t == 0 ? mt0 : mt1;
    if (mt >= 13) continue;
    for (int u = 0; u < 2; ++u) {
      int dg = dbase + u * 16 + nsel;
      float g = gamma1[dg];
      for (int r = 0; r < 8; ++r) {
        int m = mt * 16 + r + khalf * 8;
        if (m < N_) {
          int idx = (b * N_ + m) * D_ + dg;
          out[idx] = g * (acc[t][u][r] + b_attn[m]) + x[idx];
        }
      }
    }
  }
}

// ------------------------------ channel mixing ------------------------------
// Fully fused per 32-token tile (dynamic LDS): h1 = gelu(aff2(x1) @ W1^T + b1)
// kept in LDS, out = gamma2*(h1 @ W2^T + b2) + x1 in place over d_out.
// One weight fragment feeds WMMAs for both 16-row halves -> halves L2 traffic.
// grid: BN/32 = 1568; block 256 (8 waves).

__global__ __launch_bounds__(256) void k_channelmix(
    float* __restrict__ xio,
    const float* __restrict__ alpha2, const float* __restrict__ beta2,
    const _Float16* __restrict__ W1q, const float* __restrict__ b1,
    const _Float16* __restrict__ W2q, const float* __restrict__ b2,
    const float* __restrict__ gamma2)
{
  extern __shared__ char smem[];
  _Float16 (*x2s)[X2S_STRIDE] = (_Float16(*)[X2S_STRIDE])smem;                       // [32][392]
  _Float16 (*h1s)[H1S_STRIDE] = (_Float16(*)[H1S_STRIDE])(smem + CM_ROWS * X2S_STRIDE * 2);  // [32][1544]

  const int tid   = threadIdx.x;
  const int lane  = tid & 31;
  const int wid   = tid >> 5;
  const int nsel  = lane & 15;
  const int khalf = lane >> 4;
  const int row0  = blockIdx.x * CM_ROWS;

  for (int idx = tid; idx < CM_ROWS * D_; idx += 256) {
    int r = idx / D_;
    int c = idx % D_;
    float v = alpha2[c] * xio[(row0 + r) * D_ + c] + beta2[c];
    x2s[r][c] = (_Float16)v;
  }
  __syncthreads();

  // Phase A: 96 j-tiles over 8 waves, K = 384 (12 chunks of 32), 2 row-halves
  for (int jt = wid; jt < H_ / 16; jt += 8) {
    int jbase = jt * 16;
    v8f acc0 = {}; v8f acc1 = {};
    for (int kc = 0; kc < D_ / 32; ++kc) {
      const _Float16* bp = W1q + (jbase + nsel) * D_ + kc * 32 + khalf * 16;
      v16h bm = cat16(*(const v8h*)bp, *(const v8h*)(bp + 8));
      const _Float16* ap0 = &x2s[nsel][kc * 32 + khalf * 16];
      v16h am0 = cat16(*(const v8h*)ap0, *(const v8h*)(ap0 + 8));
      const _Float16* ap1 = &x2s[16 + nsel][kc * 32 + khalf * 16];
      v16h am1 = cat16(*(const v8h*)ap1, *(const v8h*)(ap1 + 8));
      acc0 = wmma_f16(am0, bm, acc0);
      acc1 = wmma_f16(am1, bm, acc1);
    }
    int j = jbase + nsel;
    float bj = b1[j];
    for (int r = 0; r < 8; ++r) {
      int rr = r + khalf * 8;
      h1s[rr][j]      = (_Float16)gelu_tanh(acc0[r] + bj);
      h1s[16 + rr][j] = (_Float16)gelu_tanh(acc1[r] + bj);
    }
  }
  __syncthreads();

  // Phase B: 24 d-tiles over 8 waves, K = 1536 (48 chunks of 32), 2 row-halves
  for (int dt = wid; dt < D_ / 16; dt += 8) {
    int dbase = dt * 16;
    v8f acc0 = {}; v8f acc1 = {};
    for (int kc = 0; kc < H_ / 32; ++kc) {
      const _Float16* bp = W2q + (dbase + nsel) * H_ + kc * 32 + khalf * 16;
      v16h bm = cat16(*(const v8h*)bp, *(const v8h*)(bp + 8));
      const _Float16* ap0 = &h1s[nsel][kc * 32 + khalf * 16];
      v16h am0 = cat16(*(const v8h*)ap0, *(const v8h*)(ap0 + 8));
      const _Float16* ap1 = &h1s[16 + nsel][kc * 32 + khalf * 16];
      v16h am1 = cat16(*(const v8h*)ap1, *(const v8h*)(ap1 + 8));
      acc0 = wmma_f16(am0, bm, acc0);
      acc1 = wmma_f16(am1, bm, acc1);
    }
    int d = dbase + nsel;
    float bd = b2[d];
    float gd = gamma2[d];
    for (int r = 0; r < 8; ++r) {
      int rowg = row0 + r + khalf * 8;
      int idx  = rowg * D_ + d;
      xio[idx] = gd * (acc0[r] + bd) + xio[idx];        // read residual, overwrite
      int idx2 = (rowg + 16) * D_ + d;
      xio[idx2] = gd * (acc1[r] + bd) + xio[idx2];
    }
  }
}

// --------------------------------- launcher ---------------------------------

extern "C" void kernel_launch(void* const* d_in, const int* in_sizes, int n_in,
                              void* d_out, int out_size, void* d_ws, size_t ws_size,
                              hipStream_t stream)
{
  (void)in_sizes; (void)n_in; (void)out_size; (void)ws_size;

  const float* x      = (const float*)d_in[0];
  const float* alpha1 = (const float*)d_in[1];
  const float* beta1  = (const float*)d_in[2];
  const float* W_attn = (const float*)d_in[3];
  const float* b_attn = (const float*)d_in[4];
  const float* gamma1 = (const float*)d_in[5];
  const float* alpha2 = (const float*)d_in[6];
  const float* beta2  = (const float*)d_in[7];
  const float* W1     = (const float*)d_in[8];
  const float* b1     = (const float*)d_in[9];
  const float* W2     = (const float*)d_in[10];
  const float* b2     = (const float*)d_in[11];
  const float* gamma2 = (const float*)d_in[12];
  const int*   bits   = (const int*)d_in[13];

  char* ws = (char*)d_ws;
  int*      absmax = (int*)ws;                                   // 3 slots
  _Float16* WattnQ = (_Float16*)(ws + 256);                      // 208*224
  _Float16* W1Q    = (_Float16*)(ws + 256 + NPAD_M * NPAD_K * 2);
  _Float16* W2Q    = W1Q + (size_t)H_ * D_;

  k_init<<<1, 32, 0, stream>>>(absmax);
  k_absmax<<<64,  256, 0, stream>>>(W_attn, N_ * N_, absmax + 0);
  k_absmax<<<256, 256, 0, stream>>>(W1,     H_ * D_, absmax + 1);
  k_absmax<<<256, 256, 0, stream>>>(W2,     D_ * H_, absmax + 2);

  k_quant<<<(NPAD_M * NPAD_K + 255) / 256, 256, 0, stream>>>(
      W_attn, WattnQ, N_, N_, NPAD_K, NPAD_M * NPAD_K, absmax + 0, bits);
  k_quant<<<(H_ * D_ + 255) / 256, 256, 0, stream>>>(
      W1, W1Q, H_, D_, D_, H_ * D_, absmax + 1, bits);
  k_quant<<<(D_ * H_ + 255) / 256, 256, 0, stream>>>(
      W2, W2Q, D_, H_, H_, D_ * H_, absmax + 2, bits);

  k_tokenmix<<<dim3(D_ / 32, B_), 256, 0, stream>>>(
      x, alpha1, beta1, WattnQ, b_attn, gamma1, (float*)d_out);

  k_channelmix<<<(B_ * N_) / CM_ROWS, 256, CM_LDS_BYTES, stream>>>(
      (float*)d_out, alpha2, beta2, W1Q, b1, W2Q, b2, gamma2);
}
